// CausalSelfAttention_62637803044935
// MI455X (gfx1250) — matrix operand order repaired and checked by reading, compiled, not run on hardware
//
#include <hip/hip_runtime.h>
#include <hip/hip_bf16.h>
#include <stdint.h>

#define D_MODEL 2048
#define NUM_HEADS 16
#define HEAD_DIM 128
#define BATCH 4
#define SEQ 2048
#define M_ROWS (BATCH * SEQ) /* 8192 */

typedef __bf16 bf16;
typedef __attribute__((ext_vector_type(16))) __bf16        v16bf;
typedef __attribute__((ext_vector_type(8)))  float         v8f;
typedef __attribute__((ext_vector_type(8)))  unsigned short v8us;
typedef __attribute__((ext_vector_type(16))) unsigned short v16us;
typedef __attribute__((ext_vector_type(8)))  unsigned int   v8u;

// ---------- helpers ----------
__device__ inline unsigned short f2bf(float f) {
  unsigned u = __builtin_bit_cast(unsigned, f);
  unsigned r = (u + 0x7FFFu + ((u >> 16) & 1u)) >> 16; // RNE
  return (unsigned short)r;
}
__device__ inline bf16 to_bf16(float f) {
  unsigned short s = f2bf(f);
  return __builtin_bit_cast(bf16, s);
}
__device__ inline v8f zero8() {
  v8f z;
#pragma unroll
  for (int r = 0; r < 8; ++r) z[r] = 0.f;
  return z;
}
// A operand (16x32 bf16). p must already include row*ld + 16*(lane>>4).
// Elements 0..15 come from p[0..15] (k = kbase..kbase+15).
__device__ inline v16bf load_a16x32(const bf16* p) {
  v8us lo = *(const v8us*)p;
  v8us hi = *(const v8us*)(p + 8);
  v16us c;
#pragma unroll
  for (int e = 0; e < 8; ++e) { c[e] = lo[e]; c[8 + e] = hi[e]; }
  return __builtin_bit_cast(v16bf, c);
}
// B operand (32x16 bf16) from column-contiguous storage.
// p must already include col*ld + 16*(lane>>4); 16 contiguous k-values.
__device__ inline v16bf load_b32x16(const bf16* p) {
  v16us c = *(const v16us*)p;
  return __builtin_bit_cast(v16bf, c);
}
__device__ inline v8f wmma_bf16(v16bf a, v16bf b, v8f c) {
  return __builtin_amdgcn_wmma_f32_16x16x32_bf16(false, a, false, b, (short)0, c,
                                                 false, false);
}

// ---------- elementwise fp32 -> bf16 ----------
__global__ void k_cvt_x(const float* __restrict__ x, bf16* __restrict__ xb, int n) {
  int i = blockIdx.x * blockDim.x + threadIdx.x;
  if (i < n) xb[i] = to_bf16(x[i]);
}

// ---------- transpose + convert: W[k][n] fp32 -> Wt[n][k] bf16 ----------
__global__ void k_transpose_cvt(const float* __restrict__ W, bf16* __restrict__ Wt) {
  __shared__ float t[32][33];
  int k0 = blockIdx.x * 32, n0 = blockIdx.y * 32;
  int tx = threadIdx.x, ty = threadIdx.y;
#pragma unroll
  for (int i = 0; i < 32; i += 8)
    t[ty + i][tx] = W[(size_t)(k0 + ty + i) * D_MODEL + (n0 + tx)];
  __syncthreads();
#pragma unroll
  for (int i = 0; i < 32; i += 8)
    Wt[(size_t)(n0 + ty + i) * D_MODEL + (k0 + tx)] = to_bf16(t[tx][ty + i]);
}

// ---------- QKV projection GEMM: C = X * Wt^T + bias ----------
// One wave computes a 64x64 tile. VT=false -> out[b][h][s][d]; VT=true -> out[b][h][d][s].
template <bool VT>
__global__ __launch_bounds__(32) void k_gemm_qkv(const bf16* __restrict__ X,
                                                 const bf16* __restrict__ Wt,
                                                 const float* __restrict__ bias,
                                                 bf16* __restrict__ out) {
  const int lane = threadIdx.x & 31;
  const int nl = lane & 15;
  const int hi = lane >> 4;
  const int m0 = blockIdx.y * 64, n0 = blockIdx.x * 64;
  const bf16* Xp = X + (size_t)m0 * D_MODEL;
  const bf16* Wp = Wt + (size_t)n0 * D_MODEL;

  v8f acc[4][4];
#pragma unroll
  for (int i = 0; i < 4; ++i)
#pragma unroll
    for (int j = 0; j < 4; ++j) acc[i][j] = zero8();

  for (int k = 0; k < D_MODEL; k += 32) {
    v16bf a[4], bm[4];
#pragma unroll
    for (int i = 0; i < 4; ++i)
      a[i] = load_a16x32(Xp + (size_t)(16 * i + nl) * D_MODEL + k + 16 * hi);
#pragma unroll
    for (int j = 0; j < 4; ++j)
      bm[j] = load_b32x16(Wp + (size_t)(16 * j + nl) * D_MODEL + k + 16 * hi);
#pragma unroll
    for (int i = 0; i < 4; ++i)
#pragma unroll
      for (int j = 0; j < 4; ++j) acc[i][j] = wmma_bf16(a[i], bm[j], acc[i][j]);
  }

#pragma unroll
  for (int i = 0; i < 4; ++i)
#pragma unroll
    for (int j = 0; j < 4; ++j) {
      const int nn = n0 + 16 * j + nl;
      const float bv = bias[nn];
      const int h = nn >> 7, d = nn & (HEAD_DIM - 1);
#pragma unroll
      for (int r = 0; r < 8; ++r) {
        const int m = m0 + 16 * i + 8 * hi + r;
        const int bb = m >> 11, s = m & (SEQ - 1);
        const float v = acc[i][j][r] + bv;
        size_t addr;
        if (VT)
          addr = ((size_t)(bb * NUM_HEADS + h) * HEAD_DIM + d) * SEQ + s;
        else
          addr = ((size_t)(bb * NUM_HEADS + h) * SEQ + s) * HEAD_DIM + d;
        out[addr] = to_bf16(v);
      }
    }
}

// ---------- output projection GEMM: fp32 out = attn * Wt^T + bias ----------
__global__ __launch_bounds__(32) void k_gemm_out(const bf16* __restrict__ X,
                                                 const bf16* __restrict__ Wt,
                                                 const float* __restrict__ bias,
                                                 float* __restrict__ out) {
  const int lane = threadIdx.x & 31;
  const int nl = lane & 15;
  const int hi = lane >> 4;
  const int m0 = blockIdx.y * 64, n0 = blockIdx.x * 64;
  const bf16* Xp = X + (size_t)m0 * D_MODEL;
  const bf16* Wp = Wt + (size_t)n0 * D_MODEL;

  v8f acc[4][4];
#pragma unroll
  for (int i = 0; i < 4; ++i)
#pragma unroll
    for (int j = 0; j < 4; ++j) acc[i][j] = zero8();

  for (int k = 0; k < D_MODEL; k += 32) {
    v16bf a[4], bm[4];
#pragma unroll
    for (int i = 0; i < 4; ++i)
      a[i] = load_a16x32(Xp + (size_t)(16 * i + nl) * D_MODEL + k + 16 * hi);
#pragma unroll
    for (int j = 0; j < 4; ++j)
      bm[j] = load_b32x16(Wp + (size_t)(16 * j + nl) * D_MODEL + k + 16 * hi);
#pragma unroll
    for (int i = 0; i < 4; ++i)
#pragma unroll
      for (int j = 0; j < 4; ++j) acc[i][j] = wmma_bf16(a[i], bm[j], acc[i][j]);
  }

#pragma unroll
  for (int i = 0; i < 4; ++i)
#pragma unroll
    for (int j = 0; j < 4; ++j) {
      const int nn = n0 + 16 * j + nl;
      const float bv = bias[nn];
#pragma unroll
      for (int r = 0; r < 8; ++r) {
        const int m = m0 + 16 * i + 8 * hi + r;
        out[(size_t)m * D_MODEL + nn] = acc[i][j][r] + bv;
      }
    }
}

// ---------- flash attention: one wave per (b, h, 16-query tile) ----------
// Computes S^T = K * Q^T (queries land one-per-lane-column), online softmax with
// per-lane scalar (max,sum), then O^T = V^T * P^T; writes merged-head bf16 output.
__global__ __launch_bounds__(32) void k_flash(const bf16* __restrict__ Q,
                                              const bf16* __restrict__ K,
                                              const bf16* __restrict__ Vt,
                                              bf16* __restrict__ attn) {
  const int lane = threadIdx.x & 31;
  const int nl = lane & 15;  // query index within tile
  const int hi = lane >> 4;  // half-wave id
  const int q0 = blockIdx.x * 16;
  const int h = blockIdx.y;
  const int b = blockIdx.z;
  const size_t headOff = (size_t)(b * NUM_HEADS + h) * SEQ * HEAD_DIM;
  const bf16* Qh = Q + headOff;   // [SEQ][HEAD_DIM]
  const bf16* Kh = K + headOff;   // [SEQ][HEAD_DIM]
  const bf16* Vh = Vt + headOff;  // [HEAD_DIM][SEQ]

  // Q^T as the B operand: column = query nl, k runs over head_dim (contiguous).
  v16bf qb[4];
#pragma unroll
  for (int s = 0; s < 4; ++s)
    qb[s] = load_b32x16(Qh + (size_t)(q0 + nl) * HEAD_DIM + s * 32 + 16 * hi);

  float mi = -3.0e38f;
  float li = 0.f;
  v8f o[8];
#pragma unroll
  for (int dt = 0; dt < 8; ++dt) o[dt] = zero8();
  const float scale = 0.088388347648318447f; // 1/sqrt(128)
  const int qg = q0 + nl;

  for (int kb = 0; kb < q0 + 16; kb += 32) {
    // S^T tiles: sa covers keys kb..kb+15, sb covers kb+16..kb+31
    v8f sa = zero8(), sb = zero8();
#pragma unroll
    for (int s = 0; s < 4; ++s) {
      v16bf ka = load_a16x32(Kh + (size_t)(kb + nl) * HEAD_DIM + s * 32 + 16 * hi);
      v16bf k2 = load_a16x32(Kh + (size_t)(kb + 16 + nl) * HEAD_DIM + s * 32 + 16 * hi);
      sa = wmma_bf16(ka, qb[s], sa);
      sb = wmma_bf16(k2, qb[s], sb);
    }
    // mask + scale; this lane holds keys kb+8*hi+r (sa) and kb+16+8*hi+r (sb)
    float pa[8], pb[8];
    float tmax = -3.0e38f;
#pragma unroll
    for (int r = 0; r < 8; ++r) {
      const int kga = kb + 8 * hi + r;
      const int kgb = kb + 16 + 8 * hi + r;
      const float va = (kga <= qg) ? sa[r] * scale : -3.0e38f;
      const float vb = (kgb <= qg) ? sb[r] * scale : -3.0e38f;
      pa[r] = va; pb[r] = vb;
      tmax = fmaxf(tmax, fmaxf(va, vb));
    }
    tmax = fmaxf(tmax, __shfl_xor(tmax, 16, 32)); // both halves of each query agree
    const float mnew = fmaxf(mi, tmax);
    const float corr = __expf(mi - mnew);
    float psum = 0.f;
#pragma unroll
    for (int r = 0; r < 8; ++r) {
      pa[r] = __expf(pa[r] - mnew);
      pb[r] = __expf(pb[r] - mnew);
      psum += pa[r] + pb[r];
    }
    psum += __shfl_xor(psum, 16, 32);
    mi = mnew;
    li = li * corr + psum;

    // Build P^T B operand: lane needs keys 16*hi .. 16*hi+15 (contiguous in e).
    unsigned pak[4], pbk[4];
#pragma unroll
    for (int r = 0; r < 4; ++r) {
      pak[r] = (unsigned)f2bf(pa[2 * r]) | ((unsigned)f2bf(pa[2 * r + 1]) << 16);
      pbk[r] = (unsigned)f2bf(pb[2 * r]) | ((unsigned)f2bf(pb[2 * r + 1]) << 16);
    }
    v8u ptb;
#pragma unroll
    for (int r = 0; r < 4; ++r) {
      const unsigned pao = (unsigned)__shfl_xor((int)pak[r], 16, 32);
      const unsigned pbo = (unsigned)__shfl_xor((int)pbk[r], 16, 32);
      // hi==0: e0-7 = own pa (keys 0-7), e8-15 = partner pa (keys 8-15)
      // hi==1: e0-7 = partner pb (keys 16-23), e8-15 = own pb (keys 24-31)
      ptb[r] = hi ? pbo : pak[r];
      ptb[4 + r] = hi ? pbk[r] : pao;
    }
    const v16bf pt = __builtin_bit_cast(v16bf, ptb);

    // O^T += V^T * P^T over 8 d-tiles; rescale running accumulator by corr.
#pragma unroll
    for (int dt = 0; dt < 8; ++dt) {
      v16bf va = load_a16x32(Vh + (size_t)(dt * 16 + nl) * SEQ + kb + 16 * hi);
      v8f oc;
#pragma unroll
      for (int r = 0; r < 8; ++r) oc[r] = o[dt][r] * corr;
      o[dt] = wmma_bf16(va, pt, oc);
    }
  }

  const float inv = 1.0f / li;
  bf16* dst = attn + (size_t)(b * SEQ + q0 + nl) * D_MODEL + h * HEAD_DIM;
#pragma unroll
  for (int dt = 0; dt < 8; ++dt)
#pragma unroll
    for (int r = 0; r < 8; ++r)
      dst[dt * 16 + 8 * hi + r] = to_bf16(o[dt][r] * inv);
}

// ---------- launcher ----------
extern "C" void kernel_launch(void* const* d_in, const int* in_sizes, int n_in,
                              void* d_out, int out_size, void* d_ws, size_t ws_size,
                              hipStream_t stream) {
  (void)in_sizes; (void)n_in; (void)out_size; (void)ws_size;
  const float* x = (const float*)d_in[0];
  const float* Wq = (const float*)d_in[1];
  const float* bq = (const float*)d_in[2];
  const float* Wk = (const float*)d_in[3];
  const float* bk = (const float*)d_in[4];
  const float* Wv = (const float*)d_in[5];
  const float* bv = (const float*)d_in[6];
  const float* Wo = (const float*)d_in[7];
  const float* bo = (const float*)d_in[8];
  float* out = (float*)d_out;

  char* ws = (char*)d_ws;
  size_t off = 0;
  bf16* xb = (bf16*)(ws + off); off += (size_t)M_ROWS * D_MODEL * 2;  // 32 MB (reused as attn)
  bf16* Wt = (bf16*)(ws + off); off += (size_t)D_MODEL * D_MODEL * 2; // 8 MB (reused 4x)
  bf16* Qb = (bf16*)(ws + off); off += (size_t)M_ROWS * D_MODEL * 2;  // 32 MB
  bf16* Kb = (bf16*)(ws + off); off += (size_t)M_ROWS * D_MODEL * 2;  // 32 MB
  bf16* Vt = (bf16*)(ws + off); off += (size_t)M_ROWS * D_MODEL * 2;  // 32 MB
  bf16* attn = xb; // xb is dead once the three QKV GEMMs have run

  const dim3 tposeB(32, 8, 1), tposeG(D_MODEL / 32, D_MODEL / 32, 1);
  const dim3 waveB(32, 1, 1);
  const dim3 gemmG(D_MODEL / 64, M_ROWS / 64, 1);

  const int ncvt = M_ROWS * D_MODEL;
  k_cvt_x<<<ncvt / 256, 256, 0, stream>>>(x, xb, ncvt);

  k_transpose_cvt<<<tposeG, tposeB, 0, stream>>>(Wq, Wt);
  k_gemm_qkv<false><<<gemmG, waveB, 0, stream>>>(xb, Wt, bq, Qb);
  k_transpose_cvt<<<tposeG, tposeB, 0, stream>>>(Wk, Wt);
  k_gemm_qkv<false><<<gemmG, waveB, 0, stream>>>(xb, Wt, bk, Kb);
  k_transpose_cvt<<<tposeG, tposeB, 0, stream>>>(Wv, Wt);
  k_gemm_qkv<true><<<gemmG, waveB, 0, stream>>>(xb, Wt, bv, Vt);

  k_flash<<<dim3(SEQ / 16, NUM_HEADS, BATCH), waveB, 0, stream>>>(Qb, Kb, Vt, attn);

  k_transpose_cvt<<<tposeG, tposeB, 0, stream>>>(Wo, Wt);
  k_gemm_out<<<gemmG, waveB, 0, stream>>>(attn, Wt, bo, out);
}
